// MoE_49589692400341
// MI455X (gfx1250) — matrix operand order repaired and checked by reading, not compiled, not run on hardware
//
#include <hip/hip_runtime.h>
#include <hip/hip_bf16.h>

// ---- types -----------------------------------------------------------------
typedef __attribute__((ext_vector_type(16))) __bf16          v16bf;
typedef __attribute__((ext_vector_type(8)))  float           v8f;
typedef __attribute__((ext_vector_type(4)))  unsigned int    uv4;
typedef __attribute__((ext_vector_type(4)))  int             iv4;
typedef __attribute__((ext_vector_type(4)))  float           fv4;
typedef __attribute__((ext_vector_type(4)))  unsigned short  us4;

#define D_MODEL 1024
#define D_FF    2048
#define N_EXP   8
#define TOPK    2
#define M_TILE  128
#define N_TILE  128
#define K_TILE  32
#define LDS_P   40   // padded LDS row stride in bf16 elems (80B, 16B-aligned)

// ---- CDNA5 async global->LDS copy ------------------------------------------
// Probe result (round 2 diagnostic): builtin exists, 4 args, param0 is
// int4 addrspace(1)* (printed as "int4 __device__*"), param1 lds side.
#if __has_builtin(__builtin_amdgcn_global_load_async_to_lds_b128)
  #define HAVE_ASYNC 1
  #define ASYNC_CP16(gp, lp) __builtin_amdgcn_global_load_async_to_lds_b128(   \
      (__attribute__((address_space(1))) iv4*)(void*)(size_t)(gp),             \
      (__attribute__((address_space(3))) iv4*)(void*)(lp), 0, 0)
  #if __has_builtin(__builtin_amdgcn_s_wait_asynccnt)
    #define WAIT_ASYNC(n) __builtin_amdgcn_s_wait_asynccnt(n)
  #else
    #define WAIT_ASYNC(n) asm volatile("s_wait_asynccnt %0" :: "i"(n) : "memory")
  #endif
#else
  #define HAVE_ASYNC 0
#endif

static __device__ __forceinline__ unsigned short f2bf(float f) {
  unsigned u = __float_as_uint(f);
  u += 0x7FFFu + ((u >> 16) & 1u);   // round-to-nearest-even
  return (unsigned short)(u >> 16);
}

// ---- init ------------------------------------------------------------------
__global__ void k_init(int* counts, int* cursor) {
  int i = threadIdx.x;
  if (i < N_EXP) { counts[i] = 0; cursor[i] = 0; }
}

// ---- f32 -> bf16 elementwise (x) ------------------------------------------
__global__ void k_cvt(const float* __restrict__ in, unsigned short* __restrict__ out, int n4) {
  int i = blockIdx.x * blockDim.x + threadIdx.x;
  if (i >= n4) return;
  fv4 v = ((const fv4*)in)[i];
  us4 o;
  o.x = f2bf(v.x); o.y = f2bf(v.y); o.z = f2bf(v.z); o.w = f2bf(v.w);
  ((us4*)out)[i] = o;
}

// ---- f32 [E][K][N] -> bf16 [E][N][K] tiled transpose-convert ---------------
__global__ void k_cvt_t(const float* __restrict__ in, unsigned short* __restrict__ out,
                        int K, int N) {
  __shared__ float tile[32][33];
  int e = blockIdx.z;
  const float* inp = in + (size_t)e * K * N;
  unsigned short* outp = out + (size_t)e * K * N;
  int n = blockIdx.y * 32 + threadIdx.x;
#pragma unroll
  for (int j = 0; j < 4; ++j) {
    int k = blockIdx.x * 32 + threadIdx.y + j * 8;
    tile[threadIdx.y + j * 8][threadIdx.x] = inp[(size_t)k * N + n];
  }
  __syncthreads();
  int kk = blockIdx.x * 32 + threadIdx.x;
#pragma unroll
  for (int j = 0; j < 4; ++j) {
    int nn = blockIdx.y * 32 + threadIdx.y + j * 8;
    outp[(size_t)nn * K + kk] = f2bf(tile[threadIdx.x][threadIdx.y + j * 8]);
  }
}

// ---- gating: one wave32 per token ------------------------------------------
__global__ void k_gate(const float* __restrict__ x, const float* __restrict__ wg,
                       int* __restrict__ top_idx, float* __restrict__ top_w,
                       int* __restrict__ counts, int T) {
  int t = (blockIdx.x * blockDim.x + threadIdx.x) >> 5;
  int lane = threadIdx.x & 31;
  if (t >= T) return;
  const float* xr = x + (size_t)t * D_MODEL;
  float acc[N_EXP];
#pragma unroll
  for (int e = 0; e < N_EXP; ++e) acc[e] = 0.f;
  for (int k = lane; k < D_MODEL; k += 32) {
    float xv = xr[k];
    const float* wr = wg + (size_t)k * N_EXP;
#pragma unroll
    for (int e = 0; e < N_EXP; ++e) acc[e] += xv * wr[e];
  }
#pragma unroll
  for (int e = 0; e < N_EXP; ++e)
    for (int off = 16; off; off >>= 1) acc[e] += __shfl_xor(acc[e], off, 32);
  if (lane == 0) {
    int i0 = 0; float v0 = acc[0];
    for (int e = 1; e < N_EXP; ++e) if (acc[e] > v0) { v0 = acc[e]; i0 = e; }
    int i1 = (i0 == 0) ? 1 : 0; float v1 = acc[i1];
    for (int e = 0; e < N_EXP; ++e) if (e != i0 && acc[e] > v1) { v1 = acc[e]; i1 = e; }
    float e1 = __expf(v1 - v0);
    float inv = 1.f / (1.f + e1);
    top_idx[2 * t] = i0;     top_w[2 * t] = inv;
    top_idx[2 * t + 1] = i1; top_w[2 * t + 1] = e1 * inv;
    atomicAdd(&counts[i0], 1);
    atomicAdd(&counts[i1], 1);
  }
}

// ---- exclusive prefix over 8 experts ---------------------------------------
__global__ void k_scan(const int* __restrict__ counts, int* __restrict__ offsets) {
  if (threadIdx.x == 0) {
    int off = 0;
    for (int e = 0; e < N_EXP; ++e) { offsets[e] = off; off += counts[e]; }
  }
}

// ---- build compacted per-expert token lists --------------------------------
__global__ void k_assign(const int* __restrict__ top_idx, int* __restrict__ cursor,
                         const int* __restrict__ offsets, int* __restrict__ rows,
                         int* __restrict__ slot_of, int T) {
  int t = blockIdx.x * blockDim.x + threadIdx.x;
  if (t >= T) return;
#pragma unroll
  for (int j = 0; j < TOPK; ++j) {
    int e = top_idx[2 * t + j];
    int p = atomicAdd(&cursor[e], 1);
    int s = offsets[e] + p;
    rows[s] = t;
    slot_of[2 * t + j] = s;
  }
}

// ---- WMMA fragment helper --------------------------------------------------
union Frag16 { uv4 q[2]; v16bf v; };

// ---- fused expert GEMM: 128x128 tile, 8-wave WG, each wave 32x64 outputs ---
// UP:  A = xb gathered via rows[] (stride KDIM), gelu, bf16 out (stride NOUT)
// !UP: A = hb contiguous slots (stride KDIM), f32 out (stride NOUT)
template <int KDIM, int NOUT, bool UP>
__global__ __launch_bounds__(256) void k_ffn(
    const unsigned short* __restrict__ A,
    const unsigned short* __restrict__ Bw,   // [E][NOUT][KDIM] bf16, N-major
    unsigned short* __restrict__ out_bf,
    float* __restrict__ out_f,
    const int* __restrict__ rows, const int* __restrict__ counts,
    const int* __restrict__ offsets, int mtiles)
{
  __shared__ __align__(16) unsigned short As[2][M_TILE][LDS_P];
  __shared__ __align__(16) unsigned short Bs[2][N_TILE][LDS_P];
  __shared__ int tokrow[M_TILE];

  int e  = blockIdx.x / mtiles;
  int mt = blockIdx.x % mtiles;
  int cnt = counts[e];
  int m0 = mt * M_TILE;
  if (m0 >= cnt) return;
  int sbase = offsets[e];
  int n0 = blockIdx.y * N_TILE;
  int tid = threadIdx.x;

  if constexpr (UP) {
    if (tid < M_TILE) {
      int m = m0 + tid;
      tokrow[tid] = rows[sbase + ((m < cnt) ? m : 0)];
    }
    __syncthreads();
  }

  int lane = tid & 31, wv = tid >> 5;
  int mstrip = (wv & 3) * 32, nbase = (wv >> 2) * 64;
  int hi = lane >> 4, lm = lane & 15;

  int sr = tid >> 1, sc = (tid & 1) * 16;   // stage: 128 rows x 32 k, 16 elems/thr

  const unsigned short* arow;
  if constexpr (UP) {
    arow = A + (size_t)tokrow[sr] * KDIM + sc;
  } else {
    int arid = m0 + sr; if (arid >= cnt) arid = cnt - 1;
    arow = A + (size_t)(sbase + arid) * KDIM + sc;
  }
  const unsigned short* brow = Bw + (size_t)e * NOUT * KDIM
                             + (size_t)(n0 + sr) * KDIM + sc;

  auto stage = [&](int buf, int k0) {
#if HAVE_ASYNC
    ASYNC_CP16(arow + k0,     &As[buf][sr][sc]);
    ASYNC_CP16(arow + k0 + 8, &As[buf][sr][sc + 8]);
    ASYNC_CP16(brow + k0,     &Bs[buf][sr][sc]);
    ASYNC_CP16(brow + k0 + 8, &Bs[buf][sr][sc + 8]);
#else
    *(uv4*)&As[buf][sr][sc]     = *(const uv4*)(arow + k0);
    *(uv4*)&As[buf][sr][sc + 8] = *(const uv4*)(arow + k0 + 8);
    *(uv4*)&Bs[buf][sr][sc]     = *(const uv4*)(brow + k0);
    *(uv4*)&Bs[buf][sr][sc + 8] = *(const uv4*)(brow + k0 + 8);
#endif
  };

  v8f acc[2][4] = {};
  stage(0, 0);
  for (int k0 = 0; k0 < KDIM; k0 += K_TILE) {
    int buf = (k0 / K_TILE) & 1;
    bool more = (k0 + K_TILE) < KDIM;
    if (more) stage(buf ^ 1, k0 + K_TILE);   // fill other buffer while computing
#if HAVE_ASYNC
    if (more) { WAIT_ASYNC(4); } else { WAIT_ASYNC(0); }  // this tile landed
#else
    if (more) {
      __builtin_prefetch(arow + k0 + 2 * K_TILE);
      __builtin_prefetch(brow + k0 + 2 * K_TILE);
    }
#endif
    __syncthreads();                          // all waves' tile visible

    Frag16 af0, af1;
    af0.q[0] = *(const uv4*)&As[buf][mstrip + lm][hi * 8];
    af0.q[1] = *(const uv4*)&As[buf][mstrip + lm][hi * 8 + 16];
    af1.q[0] = *(const uv4*)&As[buf][mstrip + 16 + lm][hi * 8];
    af1.q[1] = *(const uv4*)&As[buf][mstrip + 16 + lm][hi * 8 + 16];
#pragma unroll
    for (int nb = 0; nb < 4; ++nb) {
      Frag16 bf;
      int nn = nbase + nb * 16 + lm;
      bf.q[0] = *(const uv4*)&Bs[buf][nn][hi * 8];
      bf.q[1] = *(const uv4*)&Bs[buf][nn][hi * 8 + 16];
      acc[0][nb] = __builtin_amdgcn_wmma_f32_16x16x32_bf16(
          false, af0.v, false, bf.v, (short)0, acc[0][nb], false, false);
      acc[1][nb] = __builtin_amdgcn_wmma_f32_16x16x32_bf16(
          false, af1.v, false, bf.v, (short)0, acc[1][nb], false, false);
    }
    __syncthreads();                          // compute done; buffer reusable
  }

#pragma unroll
  for (int half = 0; half < 2; ++half) {
#pragma unroll
    for (int nb = 0; nb < 4; ++nb) {
      int nn = n0 + nbase + nb * 16 + lm;
#pragma unroll
      for (int v = 0; v < 8; ++v) {
        int ml = mstrip + half * 16 + hi * 8 + v;
        int m = m0 + ml;
        if (m < cnt) {
          float tv = acc[half][nb][v];
          if constexpr (UP) {
            float g = 0.5f * tv * (1.0f + erff(tv * 0.70710678118654752f));
            out_bf[(size_t)(sbase + m) * NOUT + nn] = f2bf(g);
          } else {
            out_f[(size_t)(sbase + m) * NOUT + nn] = tv;
          }
        }
      }
    }
  }
}

// ---- weighted top-2 combine ------------------------------------------------
__global__ void k_combine(const fv4* __restrict__ y, const int* __restrict__ slot_of,
                          const float* __restrict__ top_w, fv4* __restrict__ out) {
  int t = blockIdx.x;
  int c = threadIdx.x;                       // 256 fv4 = 1024 floats per token
  int s0 = slot_of[2 * t], s1 = slot_of[2 * t + 1];
  float w0 = top_w[2 * t], w1 = top_w[2 * t + 1];
  fv4 a = y[(size_t)s0 * (D_MODEL / 4) + c];
  fv4 b = y[(size_t)s1 * (D_MODEL / 4) + c];
  fv4 o;
  o.x = w0 * a.x + w1 * b.x;
  o.y = w0 * a.y + w1 * b.y;
  o.z = w0 * a.z + w1 * b.z;
  o.w = w0 * a.w + w1 * b.w;
  out[(size_t)t * (D_MODEL / 4) + c] = o;
}

// ---- host ------------------------------------------------------------------
extern "C" void kernel_launch(void* const* d_in, const int* in_sizes, int n_in,
                              void* d_out, int out_size, void* d_ws, size_t ws_size,
                              hipStream_t stream) {
  const float* x  = (const float*)d_in[0];
  const float* wg = (const float*)d_in[1];
  const float* w1 = (const float*)d_in[2];
  const float* w2 = (const float*)d_in[3];
  float* out = (float*)d_out;
  const int T = in_sizes[0] / D_MODEL;

  char* ws = (char*)d_ws;
  size_t cur = 0;
  auto take = [&](size_t bytes) -> void* {
    void* p = (void*)(ws + cur);
    cur += (bytes + 255) & ~(size_t)255;
    return p;
  };
  int*   counts  = (int*)take(N_EXP * 4);
  int*   cursor  = (int*)take(N_EXP * 4);
  int*   offsets = (int*)take(N_EXP * 4);
  int*   top_idx = (int*)take((size_t)2 * T * 4);
  float* top_w   = (float*)take((size_t)2 * T * 4);
  int*   rowsA   = (int*)take((size_t)2 * T * 4);
  int*   slot_of = (int*)take((size_t)2 * T * 4);
  unsigned short* xb   = (unsigned short*)take((size_t)T * D_MODEL * 2);
  unsigned short* w1t  = (unsigned short*)take((size_t)N_EXP * D_MODEL * D_FF * 2);
  unsigned short* w2t  = (unsigned short*)take((size_t)N_EXP * D_MODEL * D_FF * 2);
  unsigned short* hbuf = (unsigned short*)take((size_t)2 * T * D_FF * 2);
  float* ybuf          = (float*)take((size_t)2 * T * D_MODEL * 4);
  (void)ws_size; (void)n_in; (void)out_size;

  const int mtiles = (T + M_TILE - 1) / M_TILE;

  k_init<<<1, 64, 0, stream>>>(counts, cursor);
  int n4x = T * D_MODEL / 4;
  k_cvt<<<(n4x + 255) / 256, 256, 0, stream>>>(x, xb, n4x);
  k_cvt_t<<<dim3(D_MODEL / 32, D_FF / 32, N_EXP), dim3(32, 8), 0, stream>>>(w1, w1t, D_MODEL, D_FF);
  k_cvt_t<<<dim3(D_FF / 32, D_MODEL / 32, N_EXP), dim3(32, 8), 0, stream>>>(w2, w2t, D_FF, D_MODEL);
  k_gate<<<(T + 7) / 8, 256, 0, stream>>>(x, wg, top_idx, top_w, counts, T);
  k_scan<<<1, 32, 0, stream>>>(counts, offsets);
  k_assign<<<(T + 255) / 256, 256, 0, stream>>>(top_idx, cursor, offsets, rowsA, slot_of, T);
  k_ffn<D_MODEL, D_FF, true><<<dim3(N_EXP * mtiles, D_FF / N_TILE), 256, 0, stream>>>(
      xb, w1t, hbuf, (float*)nullptr, rowsA, counts, offsets, mtiles);
  k_ffn<D_FF, D_MODEL, false><<<dim3(N_EXP * mtiles, D_MODEL / N_TILE), 256, 0, stream>>>(
      hbuf, w2t, (unsigned short*)nullptr, ybuf, rowsA, counts, offsets, mtiles);
  k_combine<<<T, 256, 0, stream>>>((const fv4*)ybuf, slot_of, top_w, (fv4*)out);
}
